// PointNet_87110526698070
// MI455X (gfx1250) — matrix operand order repaired and checked by reading, not compiled, not run on hardware
//
#include <hip/hip_runtime.h>
#include <hip/hip_bf16.h>

typedef _Float16 v16h __attribute__((ext_vector_type(16)));
typedef float    v8f  __attribute__((ext_vector_type(8)));

// ---- workspace layout (float offsets) ----
#define WS_S      0        // 64*64   per-batch moments S[b][a*10+k], dwsum at [60]
#define WS_AGG    4096     // 64*64
#define WS_H1     8192     // 64*64
#define WS_ENC    12288    // 64*256
#define WS_Z      28672    // 64*128
#define WS_C2     36864    // 64*64   center2
#define WS_WW     40960    // 64*6*3
#define WS_ZT     42112    // 64*1536
#define WS_MEAN   140416   // 64
#define WS_ISTD   140480   // 64
#define WS_PSUM   140544   // 1024*64
#define WS_PSQ    206080   // 1024*64
#define WS_U16    271616   // _Float16[64*64*32] lives here (as halfs)

#define FENCE() do { __threadfence(); __syncthreads(); } while (0)

// ============================================================
// Kernel A: per-batch streaming reduction over N=2048 points.
// S[b][a][k] = sum_n dw_n * cos2_n[a] * dir_in_n[k]; S[b][60] = sum_n dw_n
// ============================================================
__global__ void ka_moments(const float* __restrict__ x, float* __restrict__ S)
{
    __shared__ float wacc[8][61];
    int t = threadIdx.x, b = blockIdx.x;
    int w = t >> 5, lane = t & 31;

    float Sl[61];
    #pragma unroll
    for (int i = 0; i < 61; i++) Sl[i] = 0.f;

    for (int n = t; n < 2048; n += 256) {
        const float* xp = x + ((size_t)(b * 2048 + n)) * 10;
        float xv[10];
        #pragma unroll
        for (int k = 0; k < 10; k++) xv[k] = xp[k];
        float xx = xv[0], yy = xv[1], zz = xv[2];
        float n2 = xx * xx + yy * yy + zz * zz;
        float nrm = sqrtf(n2);
        float dw = 1.f - (n2 - 1.f) * (1.f / 3.f);   // inner=1, outer=2
        dw = fmaxf(dw, 0.f);
        if (nrm <= 0.f) dw = 0.f;
        float inv = 1.f / (nrm + 1e-8f);
        float c2[6], cp;
        cp = fmaxf( zz, 0.f) * inv; c2[0] = cp * cp;
        cp = fmaxf(-zz, 0.f) * inv; c2[1] = cp * cp;
        cp = fmaxf( yy, 0.f) * inv; c2[2] = cp * cp;
        cp = fmaxf(-yy, 0.f) * inv; c2[3] = cp * cp;
        cp = fmaxf( xx, 0.f) * inv; c2[4] = cp * cp;
        cp = fmaxf(-xx, 0.f) * inv; c2[5] = cp * cp;
        float din[10];
        #pragma unroll
        for (int k = 0; k < 7; k++) din[k] = xv[3 + k];
        din[7] = xx; din[8] = yy; din[9] = zz;
        #pragma unroll
        for (int a = 0; a < 6; a++) {
            float wa = dw * c2[a];
            #pragma unroll
            for (int k = 0; k < 10; k++) Sl[a * 10 + k] += wa * din[k];
        }
        Sl[60] += dw;
    }
    // wave32 reduction, then fixed-order cross-wave sum (deterministic)
    for (int i = 0; i < 61; i++) {
        float v = Sl[i];
        v += __shfl_down(v, 16, 32);
        v += __shfl_down(v, 8, 32);
        v += __shfl_down(v, 4, 32);
        v += __shfl_down(v, 2, 32);
        v += __shfl_down(v, 1, 32);
        if (lane == 0) wacc[w][i] = v;
    }
    __syncthreads();
    if (t < 61) {
        float s = 0.f;
        #pragma unroll
        for (int w2 = 0; w2 < 8; w2++) s += wacc[w2][t];
        S[b * 64 + t] = s;
    }
}

// ============================================================
// Kernel B: all small dense algebra (single 256-thread block).
// ============================================================
__global__ void kb_dense(const float* __restrict__ x,
                         const float* __restrict__ Wdir,
                         const float* __restrict__ g1, const float* __restrict__ b1,
                         const float* __restrict__ Wdir2,
                         const float* __restrict__ g2, const float* __restrict__ b2,
                         const float* __restrict__ Wf1, const float* __restrict__ bf1,
                         const float* __restrict__ Wcm, const float* __restrict__ bcm,
                         const float* __restrict__ Wdc, const float* __restrict__ Wm0,
                         float* __restrict__ ws)
{
    float* S      = ws + WS_S;
    float* agg    = ws + WS_AGG;
    float* h1     = ws + WS_H1;
    float* enc    = ws + WS_ENC;
    float* z      = ws + WS_Z;
    float* cen2   = ws + WS_C2;
    float* WWb    = ws + WS_WW;
    float* zt     = ws + WS_ZT;
    _Float16* U   = (_Float16*)(ws + WS_U16);

    __shared__ float mn[256], is[256];
    int t = threadIdx.x;

    // B1: agg[b][m] = (sum_{a,k} S[b][a,k] * Wdir[m*6+a][k]) / dwsum[b]
    for (int i = t; i < 4096; i += 256) {
        int b = i >> 6, m = i & 63;
        const float* Sb = S + b * 64;
        float acc = 0.f;
        for (int a = 0; a < 6; a++) {
            const float* wr = Wdir + (m * 6 + a) * 10;
            #pragma unroll
            for (int k = 0; k < 10; k++) acc += Sb[a * 10 + k] * wr[k];
        }
        agg[i] = acc / Sb[60];
    }
    FENCE();
    // B2: batchnorm over b (64 rows), relu -> h1
    if (t < 64) {
        float s = 0.f, q = 0.f;
        for (int b = 0; b < 64; b++) { float v = agg[b * 64 + t]; s += v; q += v * v; }
        float m = s * (1.f / 64.f), v = q * (1.f / 64.f) - m * m;
        mn[t] = m; is[t] = rsqrtf(v + 1e-5f);
    }
    __syncthreads();
    for (int i = t; i < 4096; i += 256) {
        int m = i & 63;
        h1[i] = fmaxf((agg[i] - mn[m]) * is[m] * g1[m] + b1[m], 0.f);
    }
    FENCE();
    // B3: enc_pre = h1 @ Wdir2.T  (64x256)
    for (int i = t; i < 16384; i += 256) {
        int b = i >> 8, e = i & 255;
        const float* hb = h1 + b * 64;
        const float* wr = Wdir2 + e * 64;
        float acc = 0.f;
        for (int m = 0; m < 64; m++) acc += hb[m] * wr[m];
        enc[i] = acc;
    }
    FENCE();
    // bn2 + relu (in place)
    if (t < 256) {
        float s = 0.f, q = 0.f;
        for (int b = 0; b < 64; b++) { float v = enc[b * 256 + t]; s += v; q += v * v; }
        float m = s * (1.f / 64.f), v = q * (1.f / 64.f) - m * m;
        mn[t] = m; is[t] = rsqrtf(v + 1e-5f);
    }
    __syncthreads();
    for (int i = t; i < 16384; i += 256) {
        int e = i & 255;
        enc[i] = fmaxf((enc[i] - mn[e]) * is[e] * g2[e] + b2[e], 0.f);
    }
    FENCE();
    // B4: z = enc @ Wf1.T + bf1  (64x128)
    for (int i = t; i < 8192; i += 256) {
        int b = i >> 7, v = i & 127;
        const float* eb = enc + b * 256;
        const float* wr = Wf1 + v * 256;
        float acc = bf1[v];
        for (int e = 0; e < 256; e++) acc += eb[e] * wr[e];
        z[i] = acc;
    }
    FENCE();
    // B5: center2[b][m] = [z_b, xyz0_b] @ Wcm[m] + bcm[m]
    for (int i = t; i < 4096; i += 256) {
        int b = i >> 6, m = i & 63;
        const float* zb = z + b * 128;
        const float* wr = Wcm + m * 131;
        float acc = bcm[m];
        for (int v = 0; v < 128; v++) acc += zb[v] * wr[v];
        const float* x0 = x + (size_t)b * 2048 * 10;
        acc += x0[0] * wr[128] + x0[1] * wr[129] + x0[2] * wr[130];
        cen2[i] = acc;
    }
    // B6: zt[b][j] = z_b . Wdc[j][3:131]  (64x1536)
    for (int i = t; i < 98304; i += 256) {
        int b = i / 1536, j = i % 1536;
        const float* zb = z + b * 128;
        const float* wr = Wdc + j * 131 + 3;
        float acc = 0.f;
        for (int v = 0; v < 128; v++) acc += zb[v] * wr[v];
        zt[i] = acc;
    }
    // B7a: WW[m][a][k] = sum_e Wm0[m][e] * Wdc[e*6+a][k], k<3
    for (int i = t; i < 1152; i += 256) {
        int m = i / 18, r = i % 18, a = r / 3, k = r % 3;
        const float* w0 = Wm0 + m * 256;
        float acc = 0.f;
        for (int e = 0; e < 256; e++) acc += w0[e] * Wdc[(e * 6 + a) * 131 + k];
        WWb[i] = acc;
    }
    FENCE();
    // B7b: build f16 U[b][m][kk], kk=a*4+k : k<3 -> WW, k==3 -> ZZ[b][m][a]
    for (int i = t; i < 24576; i += 256) {
        int b = i / 384, rem = i % 384, m = rem / 6, a = rem % 6;
        const float* w0  = Wm0 + m * 256;
        const float* ztb = zt + b * 1536;
        float zzv = 0.f;
        for (int e = 0; e < 256; e++) zzv += w0[e] * ztb[e * 6 + a];
        _Float16* ub = U + ((size_t)(b * 64 + m)) * 32;
        ub[a * 4 + 0] = (_Float16)WWb[(m * 6 + a) * 3 + 0];
        ub[a * 4 + 1] = (_Float16)WWb[(m * 6 + a) * 3 + 1];
        ub[a * 4 + 2] = (_Float16)WWb[(m * 6 + a) * 3 + 2];
        ub[a * 4 + 3] = (_Float16)zzv;
    }
    // zero K-padding 24..31
    for (int i = t; i < 32768; i += 256) {
        int bm = i >> 3, kk = 24 + (i & 7);
        U[(size_t)bm * 32 + kk] = (_Float16)0.f;
    }
}

// ============================================================
// WMMA sig tile: one wave computes a 16-point x 64-channel strip.
// A: 16x32 f16 features (ISA 7.12.2 interleaved layout),
// B: 32x16 f16 U-tile (same interleaved K layout as A), 4x V_WMMA_F32_16X16X32_F16.
// ============================================================
__device__ __forceinline__ void sig_tiles(const float* __restrict__ x,
                                          const _Float16* __restrict__ U,
                                          int b, int n0, int lane, v8f* d)
{
    int p = lane & 15;
    int n = n0 + p;
    const float* xp = x + ((size_t)(b * 2048 + n)) * 10;
    float xx = xp[0], yy = xp[1], zz = xp[2];
    float nrm = sqrtf(xx * xx + yy * yy + zz * zz);
    float inv = 1.f / (nrm + 1e-8f);
    float c2[6], cp;
    cp = fmaxf( zz, 0.f) * inv; c2[0] = cp * cp;
    cp = fmaxf(-zz, 0.f) * inv; c2[1] = cp * cp;
    cp = fmaxf( yy, 0.f) * inv; c2[2] = cp * cp;
    cp = fmaxf(-yy, 0.f) * inv; c2[3] = cp * cp;
    cp = fmaxf( xx, 0.f) * inv; c2[4] = cp * cp;
    cp = fmaxf(-xx, 0.f) * inv; c2[5] = cp * cp;
    float f[24];
    #pragma unroll
    for (int a = 0; a < 6; a++) {
        f[a * 4 + 0] = c2[a] * xx;
        f[a * 4 + 1] = c2[a] * yy;
        f[a * 4 + 2] = c2[a] * zz;
        f[a * 4 + 3] = c2[a];
    }
    // A operand: lanes 0-15 hold K {0..7,16..23}, lanes 16-31 hold K {8..15,24..31}
    v16h av;
    int lo = (lane < 16) ? 0 : 8;
    #pragma unroll
    for (int i = 0; i < 16; i++) {
        int K = i + lo + ((i >= 8) ? 8 : 0);
        av[i] = (K < 24) ? (_Float16)f[K] : (_Float16)0.f;
    }
    int kb0 = (lane < 16) ? 0 : 8;    // B operand: same interleaved K layout as A
    #pragma unroll
    for (int ct = 0; ct < 4; ct++) {
        int c = ct * 16 + p;
        const _Float16* u = U + ((size_t)(b * 64 + c)) * 32;
        v16h bv;
        #pragma unroll
        for (int i = 0; i < 16; i++) bv[i] = u[kb0 + i + ((i >= 8) ? 8 : 0)];
        v8f acc = {};
        d[ct] = __builtin_amdgcn_wmma_f32_16x16x32_f16(
            false, av, false, bv, (short)0, acc, false, false);
    }
}

// ============================================================
// Kernel C1: sig via WMMA + deterministic per-column sum/sumsq.
// ============================================================
__global__ void kc1_stats(const float* __restrict__ x, const _Float16* __restrict__ U,
                          const float* __restrict__ cen2,
                          float* __restrict__ psum, float* __restrict__ psq)
{
    __shared__ float ss[8][64];
    __shared__ float sq[8][64];
    int t = threadIdx.x, w = t >> 5, lane = t & 31, p = lane & 15;
    int b = blockIdx.x >> 4, blk = blockIdx.x & 15;
    int n0 = blk * 128 + w * 16;

    v8f d[4];
    sig_tiles(x, U, b, n0, lane, d);

    #pragma unroll
    for (int ct = 0; ct < 4; ct++) {
        if (n0 == 0 && lane < 16) d[ct][0] = cen2[b * 64 + ct * 16 + p];  // row n==0 override
        float s = 0.f, q = 0.f;
        #pragma unroll
        for (int r = 0; r < 8; r++) { float v = d[ct][r]; s += v; q += v * v; }
        s += __shfl_down(s, 16, 32);
        q += __shfl_down(q, 16, 32);
        if (lane < 16) { ss[w][ct * 16 + p] = s; sq[w][ct * 16 + p] = q; }
    }
    __syncthreads();
    if (t < 64) {
        float s = 0.f, q = 0.f;
        #pragma unroll
        for (int w2 = 0; w2 < 8; w2++) { s += ss[w2][t]; q += sq[w2][t]; }
        psum[(size_t)blockIdx.x * 64 + t] = s;
        psq [(size_t)blockIdx.x * 64 + t] = q;
    }
}

// ============================================================
// Kernel D: finalize batchnorm-3 statistics (deterministic order).
// ============================================================
__global__ void kd_finalize(const float* __restrict__ psum, const float* __restrict__ psq,
                            float* __restrict__ meanv, float* __restrict__ istdv)
{
    int c = threadIdx.x;
    if (c < 64) {
        float s = 0.f, q = 0.f;
        for (int blk = 0; blk < 1024; blk++) { s += psum[(size_t)blk * 64 + c]; q += psq[(size_t)blk * 64 + c]; }
        float m = s * (1.f / 131072.f);
        float v = q * (1.f / 131072.f) - m * m;
        meanv[c] = m;
        istdv[c] = rsqrtf(v + 1e-5f);
    }
}

// ============================================================
// Kernel C2: recompute sig via WMMA, BN+ReLU, 64->7 GEMV, sigmoid.
// ============================================================
__global__ void kc2_out(const float* __restrict__ x, const _Float16* __restrict__ U,
                        const float* __restrict__ cen2,
                        const float* __restrict__ meanv, const float* __restrict__ istdv,
                        const float* __restrict__ g3, const float* __restrict__ b3,
                        const float* __restrict__ Wf2, const float* __restrict__ bf2,
                        float* __restrict__ out)
{
    __shared__ float hbuf[128 * 64];
    __shared__ float mn[64], is[64], gg[64], bb[64];
    __shared__ float wf2[448], bff[8];
    int t = threadIdx.x;
    if (t < 64) { mn[t] = meanv[t]; is[t] = istdv[t]; gg[t] = g3[t]; bb[t] = b3[t]; }
    for (int i = t; i < 448; i += 256) wf2[i] = Wf2[i];
    if (t < 7) bff[t] = bf2[t];
    __syncthreads();

    int w = t >> 5, lane = t & 31, p = lane & 15;
    int b = blockIdx.x >> 4, blk = blockIdx.x & 15;
    int n0 = blk * 128 + w * 16;

    v8f d[4];
    sig_tiles(x, U, b, n0, lane, d);

    int mbase = (lane < 16) ? 0 : 8;   // D layout: VGPR r -> row M=r (+8 for hi lanes)
    #pragma unroll
    for (int ct = 0; ct < 4; ct++) {
        if (n0 == 0 && lane < 16) d[ct][0] = cen2[b * 64 + ct * 16 + p];
        int c = ct * 16 + p;
        #pragma unroll
        for (int r = 0; r < 8; r++) {
            float v = (d[ct][r] - mn[c]) * is[c] * gg[c] + bb[c];
            v = fmaxf(v, 0.f);
            int pl = w * 16 + r + mbase;
            hbuf[pl * 64 + c] = v;
        }
    }
    __syncthreads();
    if (t < 128) {
        int n = blk * 128 + t;
        const float* h = hbuf + t * 64;
        float acc[7];
        #pragma unroll
        for (int j = 0; j < 7; j++) acc[j] = bff[j];
        for (int c = 0; c < 64; c++) {
            float hv = h[c];
            #pragma unroll
            for (int j = 0; j < 7; j++) acc[j] += hv * wf2[j * 64 + c];
        }
        float* op = out + ((size_t)(b * 2048 + n)) * 7;
        #pragma unroll
        for (int j = 0; j < 7; j++) op[j] = 1.f / (1.f + expf(-acc[j]));
    }
}

// ============================================================
extern "C" void kernel_launch(void* const* d_in, const int* in_sizes, int n_in,
                              void* d_out, int out_size, void* d_ws, size_t ws_size,
                              hipStream_t stream)
{
    const float* x     = (const float*)d_in[0];
    // d_in[1]=Wc, d_in[2]=bc are dead code in the reference (center unused)
    const float* Wdir  = (const float*)d_in[3];
    const float* g1    = (const float*)d_in[4];
    const float* b1    = (const float*)d_in[5];
    const float* Wdir2 = (const float*)d_in[6];
    const float* g2    = (const float*)d_in[7];
    const float* b2    = (const float*)d_in[8];
    const float* Wf1   = (const float*)d_in[9];
    const float* bf1   = (const float*)d_in[10];
    const float* Wcm   = (const float*)d_in[11];
    const float* bcm   = (const float*)d_in[12];
    const float* Wdc   = (const float*)d_in[13];
    const float* Wm0   = (const float*)d_in[14];
    const float* g3    = (const float*)d_in[15];
    const float* b3    = (const float*)d_in[16];
    const float* Wf2   = (const float*)d_in[17];
    const float* bf2   = (const float*)d_in[18];

    float* ws  = (float*)d_ws;
    float* out = (float*)d_out;
    const _Float16* U = (const _Float16*)(ws + WS_U16);

    ka_moments<<<64, 256, 0, stream>>>(x, ws + WS_S);
    kb_dense<<<1, 256, 0, stream>>>(x, Wdir, g1, b1, Wdir2, g2, b2,
                                    Wf1, bf1, Wcm, bcm, Wdc, Wm0, ws);
    kc1_stats<<<1024, 256, 0, stream>>>(x, U, ws + WS_C2, ws + WS_PSUM, ws + WS_PSQ);
    kd_finalize<<<1, 64, 0, stream>>>(ws + WS_PSUM, ws + WS_PSQ, ws + WS_MEAN, ws + WS_ISTD);
    kc2_out<<<1024, 256, 0, stream>>>(x, U, ws + WS_C2, ws + WS_MEAN, ws + WS_ISTD,
                                      g3, b3, Wf2, bf2, out);
}
